// GATLayer_37374805410056
// MI455X (gfx1250) — matrix operand order repaired and checked by reading, not compiled, not run on hardware
//
#include <hip/hip_runtime.h>
#include <hip/hip_bf16.h>

typedef __attribute__((ext_vector_type(16))) __bf16    v16bf;
typedef __attribute__((ext_vector_type(8)))  float     v8f;
typedef __attribute__((ext_vector_type(4)))  unsigned  u32x4;

union FragAB {
    unsigned u[8];
    u32x4    q[2];
    v16bf    v;
};

__device__ __forceinline__ unsigned pack_bf2(float a, float b) {
    unsigned ua = __float_as_uint(a); ua += 0x7fffu + ((ua >> 16) & 1u);
    unsigned ub = __float_as_uint(b); ub += 0x7fffu + ((ub >> 16) & 1u);
    return (ua >> 16) | (ub & 0xffff0000u);
}
// order-preserving float <-> uint mapping for atomic segment-max
__device__ __forceinline__ unsigned ford(float f) {
    unsigned u = __float_as_uint(f);
    return (u & 0x80000000u) ? ~u : (u | 0x80000000u);
}
__device__ __forceinline__ float funord(unsigned u) {
    unsigned v = (u & 0x80000000u) ? (u & 0x7fffffffu) : ~u;
    return __uint_as_float(v);
}

// W[K][128] (f32, row-major) -> wT[n*K + k] (bf16), i.e. [N][K] for B-fragment loads
__global__ void cvt_wT_kernel(const float* __restrict__ W, unsigned short* __restrict__ wT, int K) {
    int idx = blockIdx.x * 256 + threadIdx.x;
    if (idx < K * 128) {
        int k = idx >> 7, n = idx & 127;
        unsigned u = __float_as_uint(W[idx]);
        u += 0x7fffu + ((u >> 16) & 1u);
        wT[(size_t)n * K + k] = (unsigned short)(u >> 16);
    }
}

// edge_attr f32 -> bf16, streaming (4 floats / thread)
__global__ void cvt_ea_kernel(const float* __restrict__ ea, unsigned short* __restrict__ eaB,
                              long n4) {
    long idx = (long)blockIdx.x * 256 + threadIdx.x;
    if (idx < n4) {
        float4 f = ((const float4*)ea)[idx];
        uint2 t2; t2.x = pack_bf2(f.x, f.y); t2.y = pack_bf2(f.z, f.w);
        ((uint2*)eaB)[idx] = t2;
    }
}

// D[M,128] = X[M,128] @ W[128,128] + bias, f32 I/O, bf16 WMMA (f32 accumulate).
// 256 threads = 8 waves; wave w owns rows [16w,16w+16), all 8 N-tiles.
// All 8 B fragments are fetched before the WMMA burst so loads overlap the MACs.
__global__ __launch_bounds__(256)
void gemm_bf16_k128(const float* __restrict__ X, const unsigned short* __restrict__ wT,
                    const float* __restrict__ bias, float* __restrict__ D, int M) {
    __shared__ unsigned short s_a[128 * 40];  // 128 rows x 32 K (bf16), stride 40 (pad)
    const int tid  = threadIdx.x;
    const int w    = tid >> 5, lane = tid & 31;
    const int half = lane >> 4, nl = lane & 15;
    const int m0   = blockIdx.x * 128;

    const v8f vzero = {0.f, 0.f, 0.f, 0.f, 0.f, 0.f, 0.f, 0.f};
    v8f acc[8];
#pragma unroll
    for (int t = 0; t < 8; ++t) acc[t] = vzero;

    const int srow = tid >> 1;      // staging: 2 threads per row
    const int sseg = tid & 1;       // 16 floats each

    for (int kc = 0; kc < 128; kc += 32) {
        int grow = m0 + srow; if (grow >= M) grow = M - 1;
        const float4* xp = (const float4*)(X + (size_t)grow * 128 + kc + sseg * 16);
#pragma unroll
        for (int j = 0; j < 4; ++j) {
            float4 f = xp[j];
            uint2 t2; t2.x = pack_bf2(f.x, f.y); t2.y = pack_bf2(f.z, f.w);
            *(uint2*)(s_a + srow * 40 + sseg * 16 + j * 4) = t2;
        }
        __syncthreads();
        // A fragment: lane (half,nl) -> row 16w+nl, K = half*16 .. half*16+15
        FragAB a;
        const unsigned short* ap = s_a + (16 * w + nl) * 40 + half * 16;
        a.q[0] = *(const u32x4*)(ap);
        a.q[1] = *(const u32x4*)(ap + 8);
        // fetch all 8 B fragments first (16 x b128 clause), then burst the WMMAs
        FragAB b[8];
#pragma unroll
        for (int t = 0; t < 8; ++t) {
            const u32x4* wp = (const u32x4*)(wT + (size_t)(16 * t + nl) * 128 + kc + half * 16);
            b[t].q[0] = wp[0]; b[t].q[1] = wp[1];
        }
#pragma unroll
        for (int t = 0; t < 8; ++t)
            acc[t] = __builtin_amdgcn_wmma_f32_16x16x32_bf16(false, a.v, false, b[t].v,
                                                             (short)0, acc[t], false, false);
        __syncthreads();
    }
#pragma unroll
    for (int t = 0; t < 8; ++t) {
        int n = 16 * t + nl;
        float bv = bias[n];
#pragma unroll
        for (int r = 0; r < 8; ++r) {
            int row = m0 + 16 * w + half * 8 + r;   // C layout: vgpr r -> M = r (+8 for hi half)
            if (row < M) D[(size_t)row * 128 + n] = acc[t][r] + bv;
        }
    }
}

// Fused: he = edge_attr@w_e (WMMA), e = hs[src]+hd[dst]+he, score = lrelu(e)·att,
// segment-max via ordered-uint atomicMax + per-dst edge counting for the CSR build.
// One block = 16 edges; wave w = N-tile w.
__global__ __launch_bounds__(256)
void edge_score_kernel(const int* __restrict__ src, const int* __restrict__ dst,
                       const unsigned short* __restrict__ eaB,
                       const unsigned short* __restrict__ weT,
                       const float* __restrict__ hs, const float* __restrict__ hd,
                       const float* __restrict__ att,
                       float* __restrict__ score, unsigned* __restrict__ m_ord,
                       int* __restrict__ counts, int E) {
    __shared__ int   s_src[16], s_dst[16];
    __shared__ float s_part[8 * 16];
    const int tid  = threadIdx.x;
    const int w    = tid >> 5, lane = tid & 31;
    const int half = lane >> 4, nl = lane & 15;
    const int e0   = blockIdx.x * 16;

    if (tid < 16)      { int e = e0 + tid;      s_src[tid]      = src[e < E ? e : E - 1]; }
    else if (tid < 32) { int e = e0 + tid - 16; s_dst[tid - 16] = dst[e < E ? e : E - 1]; }
    __syncthreads();

    const v8f vzero = {0.f, 0.f, 0.f, 0.f, 0.f, 0.f, 0.f, 0.f};
    v8f acc = vzero;
    {
        int er = e0 + nl; if (er >= E) er = E - 1;       // edge rows are consecutive
        const unsigned short* ear = eaB + (size_t)er * 64;
        // both K-chunks: fetch A0/A1/B0/B1 (b128 clause) then two WMMAs
        FragAB a0, a1, b0, b1;
        const int kb = half * 16;
        a0.q[0] = *(const u32x4*)(ear + kb);
        a0.q[1] = *(const u32x4*)(ear + kb + 8);
        a1.q[0] = *(const u32x4*)(ear + 32 + kb);
        a1.q[1] = *(const u32x4*)(ear + 32 + kb + 8);
        const unsigned short* wcol = weT + (size_t)(16 * w + nl) * 64 + half * 16;
        b0.q[0] = ((const u32x4*)wcol)[0];
        b0.q[1] = ((const u32x4*)wcol)[1];
        b1.q[0] = ((const u32x4*)(wcol + 32))[0];
        b1.q[1] = ((const u32x4*)(wcol + 32))[1];
        acc = __builtin_amdgcn_wmma_f32_16x16x32_bf16(false, a0.v, false, b0.v,
                                                      (short)0, acc, false, false);
        acc = __builtin_amdgcn_wmma_f32_16x16x32_bf16(false, a1.v, false, b1.v,
                                                      (short)0, acc, false, false);
    }
    const int   col  = 16 * w + nl;              // output channel
    const int   h    = w >> 1;                   // head = col/32
    const float attq = att[h * 32 + (w & 1) * 16 + nl];
#pragma unroll
    for (int r = 0; r < 8; ++r) {
        const int Mr = half * 8 + r;             // edge within block (C lane layout)
        const int ss = s_src[Mr], dd = s_dst[Mr];
        float v = acc[r] + hs[(size_t)ss * 128 + col] + hd[(size_t)dd * 128 + col];
        v = v > 0.f ? v : 0.2f * v;
        float p = v * attq;                      // reduce over the 16 lanes of this half
        p += __shfl_xor(p, 1, 32);
        p += __shfl_xor(p, 2, 32);
        p += __shfl_xor(p, 4, 32);
        p += __shfl_xor(p, 8, 32);
        if (nl == 0) s_part[w * 16 + Mr] = p;
    }
    __syncthreads();
    if (tid < 64) {                              // combine the two 16-col tiles per head
        int m = tid & 15, hh = tid >> 4;
        if (e0 + m < E) {
            float sc = s_part[(2 * hh) * 16 + m] + s_part[(2 * hh + 1) * 16 + m];
            score[(size_t)(e0 + m) * 4 + hh] = sc;
            atomicMax(m_ord + (size_t)s_dst[m] * 4 + hh, ford(sc));
            if (hh == 0) atomicAdd(counts + s_dst[m], 1);
        }
    }
}

__global__ void edge_softmax_kernel(float* __restrict__ score, const int* __restrict__ dst,
                                    const unsigned* __restrict__ m_ord,
                                    float* __restrict__ denom, int E) {
    int idx = blockIdx.x * 256 + threadIdx.x;
    if (idx < E * 4) {
        int e = idx >> 2, h = idx & 3;
        int d = dst[e];
        float mm = funord(m_ord[(size_t)d * 4 + h]);
        float ex = __expf(score[idx] - mm);
        score[idx] = ex;                               // overwrite in place
        atomicAdd(denom + (size_t)d * 4 + h, ex);
    }
}

// single-block exclusive scan of per-dst counts -> base[n+1], cursor copy for scatter
__global__ __launch_bounds__(1024)
void scan_kernel(const int* __restrict__ counts, int* __restrict__ base,
                 int* __restrict__ cursor, int n) {
    __shared__ int sbuf[1024];
    __shared__ int scarry;
    const int tid = threadIdx.x;
    if (tid == 0) scarry = 0;
    __syncthreads();
    for (int c0 = 0; c0 < n; c0 += 1024) {
        int i = c0 + tid;
        int v = (i < n) ? counts[i] : 0;
        sbuf[tid] = v;
        __syncthreads();
        int x = v;
#pragma unroll
        for (int off = 1; off < 1024; off <<= 1) {
            int y = (tid >= off) ? sbuf[tid - off] : 0;
            __syncthreads();
            x += y;
            sbuf[tid] = x;
            __syncthreads();
        }
        int carry = scarry;
        if (i < n) { int ex = carry + x - v; base[i] = ex; cursor[i] = ex; }
        __syncthreads();
        if (tid == 1023) scarry = carry + x;
        __syncthreads();
    }
    if (tid == 0) base[n] = scarry;
}

__global__ void scatter_kernel(const int* __restrict__ dst, int* __restrict__ cursor,
                               int* __restrict__ eids, int E) {
    int e = blockIdx.x * 256 + threadIdx.x;
    if (e < E) {
        int pos = atomicAdd(cursor + dst[e], 1);
        eids[pos] = e;
    }
}

// block-per-destination-row: msg = sum_e alpha_e * hs[src_e] (no atomics),
// fused with residual + bias + LayerNorm epilogue.
__global__ __launch_bounds__(128)
void aggregate_ln_kernel(const int* __restrict__ base, const int* __restrict__ eids,
                         const int* __restrict__ src, const float* __restrict__ hs,
                         const float* __restrict__ ex, const float* __restrict__ denom,
                         const float* __restrict__ row_x, const float* __restrict__ bias,
                         const float* __restrict__ gamma, const float* __restrict__ beta,
                         float* __restrict__ out) {
    __shared__ float sw[4], sw2[4];
    const int r = blockIdx.x, i = threadIdx.x;
    const int h = i >> 5;
    const int jb = base[r], je = base[r + 1];
    const float dn = (je > jb) ? denom[(size_t)r * 4 + h] : 1.f;
    float acc = 0.f;
    for (int j = jb; j < je; ++j) {
        int e  = eids[j];                       // uniform across block
        int ss = src[e];
        float a = ex[(size_t)e * 4 + h];
        acc += hs[(size_t)ss * 128 + i] * a;    // coalesced 512B row from L2
    }
    float v = row_x[(size_t)r * 128 + i] + acc / dn + bias[i];
    float s = v;
#pragma unroll
    for (int m = 1; m < 32; m <<= 1) s += __shfl_xor(s, m, 32);
    if ((i & 31) == 0) sw[i >> 5] = s;
    __syncthreads();
    float mu = (sw[0] + sw[1] + sw[2] + sw[3]) * (1.f / 128.f);
    float d = v - mu;
    float sq = d * d;
#pragma unroll
    for (int m = 1; m < 32; m <<= 1) sq += __shfl_xor(sq, m, 32);
    if ((i & 31) == 0) sw2[i >> 5] = sq;
    __syncthreads();
    float var = (sw2[0] + sw2[1] + sw2[2] + sw2[3]) * (1.f / 128.f);
    out[(size_t)r * 128 + i] = d * rsqrtf(var + 1e-5f) * gamma[i] + beta[i];
}

extern "C" void kernel_launch(void* const* d_in, const int* in_sizes, int n_in,
                              void* d_out, int out_size, void* d_ws, size_t ws_size,
                              hipStream_t stream) {
    const float* row_x     = (const float*)d_in[0];
    const float* token_x   = (const float*)d_in[1];
    const int*   eidx      = (const int*)d_in[2];
    const float* edge_attr = (const float*)d_in[3];
    const float* w_l  = (const float*)d_in[6];
    const float* b_l  = (const float*)d_in[7];
    const float* w_r  = (const float*)d_in[8];
    const float* b_r  = (const float*)d_in[9];
    const float* w_e  = (const float*)d_in[10];
    const float* att  = (const float*)d_in[11];
    const float* bias = (const float*)d_in[12];
    const float* ln_g = (const float*)d_in[13];
    const float* ln_b = (const float*)d_in[14];

    const int n_rows = in_sizes[0] / 128;
    const int n_tok  = in_sizes[1] / 128;
    const int E      = in_sizes[2] / 2;
    const int* src  = eidx;
    const int* dstv = eidx + E;

    float* out = (float*)d_out;

    // workspace carve (~262 MB)
    char* p = (char*)d_ws;
    auto carve = [&](size_t bytes) -> char* {
        char* q = p; p += (bytes + 255) & ~(size_t)255; return q;
    };
    float*          hs     = (float*)carve((size_t)n_tok * 128 * 4);
    float*          hd     = (float*)carve((size_t)n_rows * 128 * 4);
    float*          score  = (float*)carve((size_t)E * 4 * 4);
    unsigned short* eaB    = (unsigned short*)carve((size_t)E * 64 * 2);
    unsigned*       m_ord  = (unsigned*)carve((size_t)n_rows * 4 * 4);
    float*          denom  = (float*)carve((size_t)n_rows * 4 * 4);
    int*            counts = (int*)carve((size_t)n_rows * 4);
    int*            base   = (int*)carve((size_t)(n_rows + 1) * 4);
    int*            cursor = (int*)carve((size_t)n_rows * 4);
    int*            eids   = (int*)carve((size_t)E * 4);
    unsigned short* wlT    = (unsigned short*)carve(128 * 128 * 2);
    unsigned short* wrT    = (unsigned short*)carve(128 * 128 * 2);
    unsigned short* weT    = (unsigned short*)carve(128 * 64 * 2);

    // token_x passthrough (second output)
    hipMemcpyAsync(out + (size_t)n_rows * 128, token_x, (size_t)n_tok * 128 * 4,
                   hipMemcpyDeviceToDevice, stream);
    // zero accumulators each call (deterministic)
    hipMemsetAsync(m_ord,  0, (size_t)n_rows * 4 * 4, stream);   // 0 == ordered(-inf)
    hipMemsetAsync(denom,  0, (size_t)n_rows * 4 * 4, stream);
    hipMemsetAsync(counts, 0, (size_t)n_rows * 4, stream);

    cvt_wT_kernel<<<(128 * 128 + 255) / 256, 256, 0, stream>>>(w_l, wlT, 128);
    cvt_wT_kernel<<<(128 * 128 + 255) / 256, 256, 0, stream>>>(w_r, wrT, 128);
    cvt_wT_kernel<<<(64 * 128 + 255) / 256, 256, 0, stream>>>(w_e, weT, 64);
    {
        long n4 = (long)E * 16;  // E*64 floats / 4
        cvt_ea_kernel<<<(unsigned)((n4 + 255) / 256), 256, 0, stream>>>(edge_attr, eaB, n4);
    }

    gemm_bf16_k128<<<(n_tok + 127) / 128, 256, 0, stream>>>(token_x, wlT, b_l, hs, n_tok);
    gemm_bf16_k128<<<(n_rows + 127) / 128, 256, 0, stream>>>(row_x, wrT, b_r, hd, n_rows);

    edge_score_kernel<<<(E + 15) / 16, 256, 0, stream>>>(src, dstv, eaB, weT, hs, hd, att,
                                                         score, m_ord, counts, E);
    scan_kernel<<<1, 1024, 0, stream>>>(counts, base, cursor, n_rows);
    scatter_kernel<<<(E + 255) / 256, 256, 0, stream>>>(dstv, cursor, eids, E);
    edge_softmax_kernel<<<(E * 4 + 255) / 256, 256, 0, stream>>>(score, dstv, m_ord, denom, E);
    aggregate_ln_kernel<<<n_rows, 128, 0, stream>>>(base, eids, src, hs, score, denom,
                                                    row_x, bias, ln_g, ln_b, out);
}
